// NeRFRGBHead_55654186222230
// MI455X (gfx1250) — matrix operand order repaired and checked by reading, not compiled, not run on hardware
//
#include <hip/hip_runtime.h>

typedef __attribute__((ext_vector_type(16))) _Float16 v16h;
typedef __attribute__((ext_vector_type(8)))  _Float16 v8h;
typedef __attribute__((ext_vector_type(8)))  float    v8f;

#define NPTS   (4096 * 128)   // R*S points
#define WAVES_PER_BLOCK 4
#define TILES_PER_WAVE  8

__device__ __forceinline__ float eluf(float x) {
    return x > 0.f ? x : (__expf(x) - 1.f);
}

// A-fragment gather for V_WMMA_F32_16X16X32_F16 (16-bit A 16x32 layout):
// lane holds row (lane&15); halves 0..15 = K[2*kbase..2*kbase+15] (contiguous, matches B)
__device__ __forceinline__ v16h ldA(const _Float16* rowChunk, int kbase) {
    v8h lo = *(const v8h*)(rowChunk + 2 * kbase);
    v8h hi = *(const v8h*)(rowChunk + 2 * kbase + 8);
    return __builtin_shufflevector(lo, hi, 0,1,2,3,4,5,6,7,8,9,10,11,12,13,14,15);
}

__device__ __forceinline__ v8f wmma16(v16h a, v16h b, v8f c) {
    return __builtin_amdgcn_wmma_f32_16x16x32_f16(false, a, false, b, (short)0, c, false, false);
}

// Stage f32 weights -> padded f16 tile in LDS (whole block cooperates)
__device__ __forceinline__ void stageW(_Float16* dst, const float* src,
                                       int dstRows, int dstPitch,
                                       int srcRows, int srcCols, float scale) {
    for (int idx = threadIdx.x; idx < dstRows * dstPitch; idx += 32 * WAVES_PER_BLOCK) {
        int r = idx / dstPitch, c = idx % dstPitch;
        float v = (r < srcRows && c < srcCols) ? src[r * srcCols + c] * scale : 0.f;
        dst[idx] = (_Float16)v;
    }
}

__global__ void __launch_bounds__(32 * WAVES_PER_BLOCK)
nerf_rgb_head(const float* __restrict__ g,          // rgb_feat (P,3,35) = (P,105)
              const float* __restrict__ bw1, const float* __restrict__ bb1,
              const float* __restrict__ bw2, const float* __restrict__ bb2,
              const float* __restrict__ vw1, const float* __restrict__ vb1,
              const float* __restrict__ vw2, const float* __restrict__ vb2,
              const float* __restrict__ rw1, const float* __restrict__ rb1,
              const float* __restrict__ rw2, const float* __restrict__ rb2,
              const float* __restrict__ rw3, const float* __restrict__ rb3,
              float* __restrict__ outA,             // rgb_in  (P,3,3)
              float* __restrict__ outB)             // rgb_out (P,3)
{
    // ---- weights in LDS, shared by all 4 waves (f16, padded pitches) ----
    __shared__ __align__(32) _Float16 sWB1[64 * 128];   // 105 -> pad 128
    __shared__ __align__(32) _Float16 sWB2[32 * 64];
    __shared__ __align__(32) _Float16 sWV1[32 * 32];    // pre-scaled by 1/3
    __shared__ __align__(32) _Float16 sWV2[32 * 32];
    __shared__ __align__(32) _Float16 sWR1[32 * 96];
    __shared__ __align__(32) _Float16 sWR2[16 * 32];
    __shared__ __align__(32) _Float16 sWR3[16 * 32];    // 3x16 -> pad 16x32
    __shared__ float sB1[64], sB2[32], sV1[32], sV2[32], sR1[32], sR2[16], sR3[16];
    // ---- per-wave activation regions ----
    __shared__ __align__(32) _Float16 xbufA[WAVES_PER_BLOCK][48 * 128];
    __shared__ __align__(32) _Float16 a1A  [WAVES_PER_BLOCK][48 * 64];
    __shared__ __align__(32) _Float16 xsA  [WAVES_PER_BLOCK][48 * 32];
    __shared__ __align__(32) _Float16 tbA  [WAVES_PER_BLOCK][48 * 32];
    __shared__ __align__(32) _Float16 h2A  [WAVES_PER_BLOCK][16 * 32];

    const int tid   = threadIdx.x;
    const int wid   = tid >> 5;
    const int lane  = tid & 31;
    const int lm    = lane & 15;
    const int lhalf = lane >> 4;
    const int kb16  = lhalf * 16;   // B-fragment K offset
    const int kb8   = lhalf * 8;    // A-fragment K offset

    _Float16* xbuf = xbufA[wid];
    _Float16* a1   = a1A[wid];
    _Float16* xs   = xsA[wid];
    _Float16* tb   = tbA[wid];
    _Float16* h2   = h2A[wid];

    // ---- one-time cooperative weight staging ----
    stageW(sWB1, bw1, 64, 128, 64, 105, 1.f);
    stageW(sWB2, bw2, 32,  64, 32,  64, 1.f);
    stageW(sWV1, vw1, 32,  32, 32,  32, 1.f / 3.f);     // fold x/num_views
    stageW(sWV2, vw2, 32,  32, 32,  32, 1.f);
    stageW(sWR1, rw1, 32,  96, 32,  96, 1.f);
    stageW(sWR2, rw2, 16,  32, 16,  32, 1.f);
    stageW(sWR3, rw3, 16,  32,  3,  16, 1.f);
    if (tid < 64) sB1[tid] = bb1[tid];
    if (tid < 32) { sB2[tid] = bb2[tid]; sV1[tid] = vb1[tid]; sV2[tid] = vb2[tid]; sR1[tid] = rb1[tid]; }
    if (tid < 16) { sR2[tid] = rb2[tid]; sR3[tid] = (tid < 3) ? rb3[tid] : 0.f; }
    // per-wave constant zero pads (never overwritten in tile loop)
    for (int idx = lane; idx < 48 * 128; idx += 32) { if ((idx & 127) >= 105) xbuf[idx] = (_Float16)0.f; }
    for (int idx = lane; idx < 16 * 32;  idx += 32) { if ((idx & 31)  >= 16)  h2[idx]  = (_Float16)0.f; }
    __syncthreads();   // weights visible to all waves

    for (int tile = 0; tile < TILES_PER_WAVE; ++tile) {
        const int tileId = (blockIdx.x * WAVES_PER_BLOCK + wid) * TILES_PER_WAVE + tile;
        const long P0 = (long)tileId * 16;

        // prefetch next tile's input block (global_prefetch_b8 path)
        if (P0 + 16 < NPTS) {
            const float* nb = g + (P0 + 16) * 105;
            for (int c = lane * 32; c < 16 * 105; c += 32 * 32)
                __builtin_prefetch(nb + c, 0, 1);
        }

        // rgb_in passthrough: first 3 features of each view
        for (int idx = lane; idx < 16 * 9; idx += 32) {
            int m = idx / 9, vc = idx % 9, v = vc / 3, c = vc % 3;
            long P = P0 + m;
            outA[P * 9 + vc] = g[P * 105 + v * 35 + c];
        }

        // ---- build X rows: [mean(35) | var(35) | feat_v(35) | 0-pad] ----
        for (int idx = lane; idx < 16 * 35; idx += 32) {           // mean / var
            int m = idx / 35, f = idx % 35;
            const float* base = g + (P0 + m) * 105;
            float x0 = base[f], x1 = base[35 + f], x2 = base[70 + f];
            float mean = (x0 + x1 + x2) * (1.f / 3.f);
            float d0 = x0 - mean, d1 = x1 - mean, d2 = x2 - mean;
            float var = (d0 * d0 + d1 * d1 + d2 * d2) * (1.f / 3.f);
            _Float16 hm = (_Float16)mean, hv = (_Float16)var;
            #pragma unroll
            for (int t = 0; t < 3; ++t) {
                xbuf[(t * 16 + m) * 128 + f]      = hm;
                xbuf[(t * 16 + m) * 128 + 35 + f] = hv;
            }
        }
        for (int idx = lane; idx < 48 * 35; idx += 32) {           // per-view feats
            int rr = idx / 35, f = idx % 35;
            int t = rr >> 4, m = rr & 15;
            xbuf[rr * 128 + 70 + f] = (_Float16)g[(P0 + m) * 105 + t * 35 + f];
        }

        // ---- layer B1: 105(pad128) -> 64, elu ----
        #pragma unroll
        for (int nt = 0; nt < 4; ++nt) {
            v8f c0 = {}, c1 = {}, c2 = {};
            const _Float16* wrow = &sWB1[(nt * 16 + lm) * 128];
            #pragma unroll
            for (int kc = 0; kc < 4; ++kc) {
                v16h B = *(const v16h*)(wrow + kc * 32 + kb16);
                c0 = wmma16(ldA(&xbuf[(0 * 16 + lm) * 128 + kc * 32], kb8), B, c0);
                c1 = wmma16(ldA(&xbuf[(1 * 16 + lm) * 128 + kc * 32], kb8), B, c1);
                c2 = wmma16(ldA(&xbuf[(2 * 16 + lm) * 128 + kc * 32], kb8), B, c2);
            }
            float bias = sB1[nt * 16 + lm];
            #pragma unroll
            for (int i = 0; i < 8; ++i) {
                int m = i + lhalf * 8, col = nt * 16 + lm;
                a1[(0 * 16 + m) * 64 + col] = (_Float16)eluf(c0[i] + bias);
                a1[(1 * 16 + m) * 64 + col] = (_Float16)eluf(c1[i] + bias);
                a1[(2 * 16 + m) * 64 + col] = (_Float16)eluf(c2[i] + bias);
            }
        }

        // ---- layer B2: 64 -> 32, elu -> xs ----
        #pragma unroll
        for (int nt = 0; nt < 2; ++nt) {
            v8f c0 = {}, c1 = {}, c2 = {};
            const _Float16* wrow = &sWB2[(nt * 16 + lm) * 64];
            #pragma unroll
            for (int kc = 0; kc < 2; ++kc) {
                v16h B = *(const v16h*)(wrow + kc * 32 + kb16);
                c0 = wmma16(ldA(&a1[(0 * 16 + lm) * 64 + kc * 32], kb8), B, c0);
                c1 = wmma16(ldA(&a1[(1 * 16 + lm) * 64 + kc * 32], kb8), B, c1);
                c2 = wmma16(ldA(&a1[(2 * 16 + lm) * 64 + kc * 32], kb8), B, c2);
            }
            float bias = sB2[nt * 16 + lm];
            #pragma unroll
            for (int i = 0; i < 8; ++i) {
                int m = i + lhalf * 8, col = nt * 16 + lm;
                xs[(0 * 16 + m) * 32 + col] = (_Float16)eluf(c0[i] + bias);
                xs[(1 * 16 + m) * 32 + col] = (_Float16)eluf(c1[i] + bias);
                xs[(2 * 16 + m) * 32 + col] = (_Float16)eluf(c2[i] + bias);
            }
        }

        // ---- vis layer 1: (x/3) -> 32, elu  (1/3 folded into sWV1) ----
        #pragma unroll
        for (int nt = 0; nt < 2; ++nt) {
            v8f c0 = {}, c1 = {}, c2 = {};
            v16h B = *(const v16h*)(&sWV1[(nt * 16 + lm) * 32 + kb16]);
            c0 = wmma16(ldA(&xs[(0 * 16 + lm) * 32], kb8), B, c0);
            c1 = wmma16(ldA(&xs[(1 * 16 + lm) * 32], kb8), B, c1);
            c2 = wmma16(ldA(&xs[(2 * 16 + lm) * 32], kb8), B, c2);
            float bias = sV1[nt * 16 + lm];
            #pragma unroll
            for (int i = 0; i < 8; ++i) {
                int m = i + lhalf * 8, col = nt * 16 + lm;
                tb[(0 * 16 + m) * 32 + col] = (_Float16)eluf(c0[i] + bias);
                tb[(1 * 16 + m) * 32 + col] = (_Float16)eluf(c1[i] + bias);
                tb[(2 * 16 + m) * 32 + col] = (_Float16)eluf(c2[i] + bias);
            }
        }

        // ---- vis layer 2 + residual: xs += elu(tb @ vw2 + b) ----
        #pragma unroll
        for (int nt = 0; nt < 2; ++nt) {
            v8f c0 = {}, c1 = {}, c2 = {};
            v16h B = *(const v16h*)(&sWV2[(nt * 16 + lm) * 32 + kb16]);
            c0 = wmma16(ldA(&tb[(0 * 16 + lm) * 32], kb8), B, c0);
            c1 = wmma16(ldA(&tb[(1 * 16 + lm) * 32], kb8), B, c1);
            c2 = wmma16(ldA(&tb[(2 * 16 + lm) * 32], kb8), B, c2);
            float bias = sV2[nt * 16 + lm];
            #pragma unroll
            for (int i = 0; i < 8; ++i) {
                int m = i + lhalf * 8, col = nt * 16 + lm;
                int o0 = (0 * 16 + m) * 32 + col, o1 = (1 * 16 + m) * 32 + col, o2 = (2 * 16 + m) * 32 + col;
                xs[o0] = (_Float16)((float)xs[o0] + eluf(c0[i] + bias));
                xs[o1] = (_Float16)((float)xs[o1] + eluf(c1[i] + bias));
                xs[o2] = (_Float16)((float)xs[o2] + eluf(c2[i] + bias));
            }
        }

        // ---- ray layer 1: h(16x96, = xs reshaped) -> 32, elu -> tb rows 0..15 ----
        #pragma unroll
        for (int nt = 0; nt < 2; ++nt) {
            v8f acc = {};
            const _Float16* wrow = &sWR1[(nt * 16 + lm) * 96];
            #pragma unroll
            for (int kc = 0; kc < 3; ++kc) {
                v16h B = *(const v16h*)(wrow + kc * 32 + kb16);
                acc = wmma16(ldA(&xs[(kc * 16 + lm) * 32], kb8), B, acc);
            }
            float bias = sR1[nt * 16 + lm];
            #pragma unroll
            for (int i = 0; i < 8; ++i)
                tb[(i + lhalf * 8) * 32 + nt * 16 + lm] = (_Float16)eluf(acc[i] + bias);
        }

        // ---- ray layer 2: 32 -> 16, elu -> h2 ----
        {
            v8f acc = {};
            v16h B = *(const v16h*)(&sWR2[lm * 32 + kb16]);
            acc = wmma16(ldA(&tb[lm * 32], kb8), B, acc);
            float bias = sR2[lm];
            #pragma unroll
            for (int i = 0; i < 8; ++i)
                h2[(i + lhalf * 8) * 32 + lm] = (_Float16)eluf(acc[i] + bias);
        }

        // ---- ray layer 3: 16(pad32) -> 3, sigmoid -> rgb_out ----
        {
            v8f acc = {};
            v16h B = *(const v16h*)(&sWR3[lm * 32 + kb16]);
            acc = wmma16(ldA(&h2[lm * 32], kb8), B, acc);
            if (lm < 3) {
                float bias = sR3[lm];
                #pragma unroll
                for (int i = 0; i < 8; ++i) {
                    long P = P0 + i + lhalf * 8;
                    outB[P * 3 + lm] = 1.f / (1.f + __expf(-(acc[i] + bias)));
                }
            }
        }
    }
}

extern "C" void kernel_launch(void* const* d_in, const int* in_sizes, int n_in,
                              void* d_out, int out_size, void* d_ws, size_t ws_size,
                              hipStream_t stream) {
    const float* g   = (const float*)d_in[0];
    const float* bw1 = (const float*)d_in[1];
    const float* bb1 = (const float*)d_in[2];
    const float* bw2 = (const float*)d_in[3];
    const float* bb2 = (const float*)d_in[4];
    const float* vw1 = (const float*)d_in[5];
    const float* vb1 = (const float*)d_in[6];
    const float* vw2 = (const float*)d_in[7];
    const float* vb2 = (const float*)d_in[8];
    const float* rw1 = (const float*)d_in[9];
    const float* rb1 = (const float*)d_in[10];
    const float* rw2 = (const float*)d_in[11];
    const float* rb2 = (const float*)d_in[12];
    const float* rw3 = (const float*)d_in[13];
    const float* rb3 = (const float*)d_in[14];

    float* outA = (float*)d_out;                      // rgb_in : P*3*3
    float* outB = outA + (size_t)NPTS * 3 * 3;        // rgb_out: P*3

    const int tiles  = NPTS / 16;                                   // 32768
    const int blocks = tiles / (WAVES_PER_BLOCK * TILES_PER_WAVE);  // 1024
    nerf_rgb_head<<<dim3(blocks), dim3(32 * WAVES_PER_BLOCK), 0, stream>>>(
        g, bw1, bb1, bw2, bb2, vw1, vb1, vw2, vb2,
        rw1, rb1, rw2, rb2, rw3, rb3, outA, outB);
}
